// CEBlock_17557826306354
// MI455X (gfx1250) — matrix operand order repaired and checked by reading, not compiled, not run on hardware
//
#include <hip/hip_runtime.h>

typedef __bf16 bf16_t;
typedef __attribute__((ext_vector_type(16))) __bf16 v16bf;
typedef __attribute__((ext_vector_type(8)))  __bf16 v8bf;
typedef __attribute__((ext_vector_type(8)))  float   v8f;

#define DIMC    768
#define HEADS   12
#define HD      64
#define HIDDEN  3072
#define LT      432
#define LS      576
#define NTOK    1008
#define BATCH   8
#define KEEP    404                 // ceil(0.7 * 576)
#define NROWS2  (LT + KEEP)         // 836 pruned rows
#define NPAD    1024                // padded attention dim (mult of 64)

// ---------------------------------------------------------------------------
// bf16 WMMA GEMM:  C[M,N] (f32) = A[M,K] (bf16,row-major) @ W[N,K]^T
// REQUIRES: N % 64 == 0, K % 32 == 0 (M handled by clamp+masked store).
// Each wave computes a 2x4 grid of 16x16 tiles (32 rows x 64 cols),
// fully unconditionally (no runtime tile guards -> no reg duplication):
//   - A fragments loaded once, reused across 4 N-tiles
//   - B fragments loaded once, reused across 2 M-tiles
//   => 12 b128 loads per 8 WMMAs.
// 4 waves per block stacked along M (block covers 128 rows x 64 cols).
// Optional bias added on store. Batched via blockIdx.z with element strides.
// ---------------------------------------------------------------------------
__global__ __launch_bounds__(128)
void wmma_gemm_bf16(const bf16_t* __restrict__ A, const bf16_t* __restrict__ W,
                    float* __restrict__ C, const float* __restrict__ bias,
                    int M, int K, int lda, int ldw, int ldc,
                    long long sA, long long sW, long long sC)
{
    const int wave = threadIdx.x >> 5;
    const int lane = threadIdx.x & 31;
    const int rl   = lane & 15;     // row (A) / col (B) within tile
    const int kg   = lane >> 4;     // K-group select

    const int n0 = blockIdx.x * 64;
    const int m0 = (blockIdx.y * 8 + wave * 2) * 16;
    if (m0 >= M) return;

    A += (long long)blockIdx.z * sA;
    W += (long long)blockIdx.z * sW;
    C += (long long)blockIdx.z * sC;

    int rA0 = m0 + rl;      if (rA0 > M - 1) rA0 = M - 1;   // clamp, store masked
    int rA1 = m0 + 16 + rl; if (rA1 > M - 1) rA1 = M - 1;
    const bf16_t* a0p = A + (long long)rA0 * lda + kg * 16;
    const bf16_t* a1p = A + (long long)rA1 * lda + kg * 16;
    const bf16_t* wp  = W + (long long)(n0 + rl) * ldw + kg * 16;
    const long long wts = (long long)16 * ldw;              // B tile stride

    v8f acc0[4] = {}, acc1[4] = {};

    for (int k0 = 0; k0 < K; k0 += 32) {
        // prefetch one K-block ahead (speculative, OOB-safe)
        __builtin_prefetch((const void*)(a0p + k0 + 32), 0, 1);
        __builtin_prefetch((const void*)(wp  + k0 + 32), 0, 1);

        // A fragments: lane<16 holds K 0..15, lane>=16 holds K 16..31
        v8bf x0 = *(const v8bf*)(a0p + k0);
        v8bf x1 = *(const v8bf*)(a0p + k0 + 8);
        v8bf y0 = *(const v8bf*)(a1p + k0);
        v8bf y1 = *(const v8bf*)(a1p + k0 + 8);
        v16bf av0, av1;
#pragma unroll
        for (int i = 0; i < 8; ++i) {
            av0[i] = x0[i]; av0[i + 8] = x1[i];
            av1[i] = y0[i]; av1[i + 8] = y1[i];
        }
#pragma unroll
        for (int t = 0; t < 4; ++t) {
            // B fragment: lane<16 -> K 0..15 of column n, lane>=16 -> K 16..31
            v16bf bv = *(const v16bf*)(wp + (long long)t * wts + k0);
            acc0[t] = __builtin_amdgcn_wmma_f32_16x16x32_bf16(
                          false, av0, false, bv, (short)0, acc0[t], false, false);
            acc1[t] = __builtin_amdgcn_wmma_f32_16x16x32_bf16(
                          false, av1, false, bv, (short)0, acc1[t], false, false);
        }
    }

#pragma unroll
    for (int t = 0; t < 4; ++t) {
        const int col = n0 + t * 16 + rl;
        const float badd = bias ? bias[col] : 0.f;
#pragma unroll
        for (int r = 0; r < 8; ++r) {
            int row = m0 + kg * 8 + r;              // VGPR r -> row m0+kg*8+r
            if (row < M) C[(long long)row * ldc + col] = acc0[t][r] + badd;
            int row2 = row + 16;
            if (row2 < M) C[(long long)row2 * ldc + col] = acc1[t][r] + badd;
        }
    }
}

// ---------------------------------------------------------------------------
// Elementwise / reduction kernels
// ---------------------------------------------------------------------------
__global__ void f32_to_bf16(const float* __restrict__ s, bf16_t* __restrict__ d,
                            long long n)
{
    long long t = (long long)blockIdx.x * 256 + threadIdx.x;
    if (t < n) d[t] = (bf16_t)s[t];
}

__global__ __launch_bounds__(256)
void layernorm_bf16(const float* __restrict__ x, const float* __restrict__ w,
                    const float* __restrict__ b, bf16_t* __restrict__ out)
{
    __shared__ float red[256];
    const long long row = blockIdx.x;
    const float* xr = x + row * DIMC;

    float v3[3];
    float s = 0.f;
#pragma unroll
    for (int j = 0; j < 3; ++j) { v3[j] = xr[threadIdx.x + j * 256]; s += v3[j]; }
    red[threadIdx.x] = s; __syncthreads();
    for (int o = 128; o > 0; o >>= 1) {
        if (threadIdx.x < o) red[threadIdx.x] += red[threadIdx.x + o];
        __syncthreads();
    }
    const float mu = red[0] / DIMC; __syncthreads();

    float v = 0.f;
#pragma unroll
    for (int j = 0; j < 3; ++j) { float d = v3[j] - mu; v += d * d; }
    red[threadIdx.x] = v; __syncthreads();
    for (int o = 128; o > 0; o >>= 1) {
        if (threadIdx.x < o) red[threadIdx.x] += red[threadIdx.x + o];
        __syncthreads();
    }
    const float rstd = rsqrtf(red[0] / DIMC + 1e-5f);

    bf16_t* orow = out + row * DIMC;
#pragma unroll
    for (int j = 0; j < 3; ++j) {
        int i = threadIdx.x + j * 256;
        orow[i] = (bf16_t)((v3[j] - mu) * rstd * w[i] + b[i]);
    }
}

// qkv f32 [B*N, 3*DIM] -> Q bf16 [B,H,NTOK,64] (*0.125),
//                         K bf16 [B,H,NPAD,64] (rows >=1008 zero-filled)
__global__ void reshape_qk(const float* __restrict__ qkv, bf16_t* __restrict__ Q,
                           bf16_t* __restrict__ Kb, long long n)
{
    long long t = (long long)blockIdx.x * 256 + threadIdx.x;  // over B*H*NPAD*64
    if (t >= n) return;
    int d = (int)(t & 63); long long r = t >> 6;
    int nn = (int)(r % NPAD); long long r2 = r / NPAD;
    int h = (int)(r2 % HEADS); int bb = (int)(r2 / HEADS);
    const long long bh = (long long)bb * HEADS + h;
    if (nn < NTOK) {
        const float* src = qkv + (long long)(bb * NTOK + nn) * (3 * DIMC) + h * HD + d;
        Q[(bh * NTOK + nn) * HD + d] = (bf16_t)(src[0] * 0.125f); // q * HD^-0.5
        Kb[t] = (bf16_t)(src[DIMC]);
    } else {
        Kb[t] = (bf16_t)0.f;
    }
}

// V transposed: Vt bf16 [B,H,64,NPAD] (zero-padded cols 1008..1023)
__global__ void reshape_vt(const float* __restrict__ qkv, bf16_t* __restrict__ Vt,
                           long long n)
{
    long long t = (long long)blockIdx.x * 256 + threadIdx.x;
    if (t >= n) return;
    int nn = (int)(t & (NPAD - 1)); long long r = t >> 10;
    int d = (int)(r & 63); long long r2 = r >> 6;
    int h = (int)(r2 % HEADS); int bb = (int)(r2 / HEADS);
    float v = 0.f;
    if (nn < NTOK)
        v = qkv[(long long)(bb * NTOK + nn) * (3 * DIMC) + 2 * DIMC + h * HD + d];
    Vt[t] = (bf16_t)v;
}

// Row softmax over first NTOK cols of S (row stride NPAD) ->
// P bf16 [.,NPAD] (pad cols 0) and attn_ce f32 for q>=LT
__global__ __launch_bounds__(256)
void softmax_attn(const float* __restrict__ S, bf16_t* __restrict__ P,
                  float* __restrict__ attn_ce)
{
    __shared__ float red[256];
    const long long row = blockIdx.x;                 // (b*H + h)*NTOK + q
    int q = (int)(row % NTOK); long long r2 = row / NTOK;
    int h = (int)(r2 % HEADS); int bb = (int)(r2 / HEADS);
    const float* srow = S + row * NPAD;

    float v[4];
    float m = -3.4e38f;
#pragma unroll
    for (int j = 0; j < 4; ++j) {
        int i = threadIdx.x + j * 256;
        v[j] = (i < NTOK) ? srow[i] : -3.4e38f;
        m = fmaxf(m, v[j]);
    }
    red[threadIdx.x] = m; __syncthreads();
    for (int o = 128; o > 0; o >>= 1) {
        if (threadIdx.x < o) red[threadIdx.x] = fmaxf(red[threadIdx.x], red[threadIdx.x + o]);
        __syncthreads();
    }
    m = red[0]; __syncthreads();

    float e[4];
    float s = 0.f;
#pragma unroll
    for (int j = 0; j < 4; ++j) {
        int i = threadIdx.x + j * 256;
        e[j] = (i < NTOK) ? __expf(v[j] - m) : 0.f;
        s += e[j];
    }
    red[threadIdx.x] = s; __syncthreads();
    for (int o = 128; o > 0; o >>= 1) {
        if (threadIdx.x < o) red[threadIdx.x] += red[threadIdx.x + o];
        __syncthreads();
    }
    const float inv = 1.f / red[0];

    bf16_t* prow = P + row * NPAD;
    float* crow = (q >= LT)
        ? attn_ce + (((long long)(bb * HEADS + h) * LS + (q - LT)) * NTOK)
        : nullptr;
#pragma unroll
    for (int j = 0; j < 4; ++j) {
        int i = threadIdx.x + j * 256;                 // covers all NPAD cols
        float pv = e[j] * inv;                         // 0 in pad region
        prow[i] = (bf16_t)pv;
        if (crow && i < NTOK) crow[i] = pv;
    }
}

// O f32 [B,H,N,64] -> merged-head bf16 [B*N, 768]
__global__ void reshape_o(const float* __restrict__ O, bf16_t* __restrict__ out,
                          long long n)
{
    long long t = (long long)blockIdx.x * 256 + threadIdx.x;
    if (t >= n) return;
    int c = (int)(t % DIMC); long long r = t / DIMC;
    int nn = (int)(r % NTOK); int bb = (int)(r / NTOK);
    int h = c >> 6, d = c & 63;
    out[t] = (bf16_t)O[((long long)(bb * HEADS + h) * NTOK + nn) * HD + d];
}

__global__ void add_res(const float* __restrict__ x, const float* __restrict__ y,
                        float* __restrict__ out, long long n)
{
    long long t = (long long)blockIdx.x * 256 + threadIdx.x;
    if (t < n) out[t] = x[t] + y[t];
}

__global__ void add_inplace(float* __restrict__ x, const float* __restrict__ y,
                            long long n)
{
    long long t = (long long)blockIdx.x * 256 + threadIdx.x;
    if (t < n) x[t] += y[t];
}

// attn_t[b,j] = || mean_h attn_ce[b,h,j,:] ||_2
__global__ __launch_bounds__(256)
void attn_norm(const float* __restrict__ attn_ce, float* __restrict__ t_out)
{
    __shared__ float red[256];
    int j = blockIdx.x % LS; int bb = blockIdx.x / LS;
    float acc = 0.f;
    for (int k = threadIdx.x; k < NTOK; k += 256) {
        float s = 0.f;
        for (int h = 0; h < HEADS; ++h)
            s += attn_ce[((long long)(bb * HEADS + h) * LS + j) * NTOK + k];
        s *= (1.f / HEADS);
        acc += s * s;
    }
    red[threadIdx.x] = acc; __syncthreads();
    for (int o = 128; o > 0; o >>= 1) {
        if (threadIdx.x < o) red[threadIdx.x] += red[threadIdx.x + o];
        __syncthreads();
    }
    if (threadIdx.x == 0) t_out[bb * LS + j] = sqrtf(red[0]);
}

// Stable descending argsort by rank counting (576^2 per batch), emit indices.
__global__ __launch_bounds__(LS)
void topk_select(const float* __restrict__ t_in, const int* __restrict__ gis,
                 const int* __restrict__ git, int* __restrict__ out_git,
                 int* __restrict__ keep, int* __restrict__ removed,
                 int* __restrict__ topk_idx)
{
    __shared__ float a[LS];
    __shared__ int order[LS];
    const int bb = blockIdx.x, j = threadIdx.x;
    a[j] = t_in[bb * LS + j];
    __syncthreads();
    const float aj = a[j];
    int rank = 0;
    for (int i = 0; i < LS; ++i) {
        float ai = a[i];
        rank += (ai > aj) || (ai == aj && i < j);
    }
    order[rank] = j;
    __syncthreads();
    const int oj = order[j];
    if (j < KEEP) {
        keep[bb * KEEP + j]     = gis[bb * LS + oj];
        topk_idx[bb * KEEP + j] = oj;
    } else {
        removed[bb * (LS - KEEP) + (j - KEEP)] = gis[bb * LS + oj];
    }
    if (j < LT) out_git[bb * LT + j] = git[bb * LT + j];
}

// Build pruned token matrix (B,836,768) from x1 (B,1008,768)
__global__ __launch_bounds__(256)
void gather_tokens(const float* __restrict__ x1, const int* __restrict__ topk_idx,
                   float* __restrict__ xo)
{
    const int row = blockIdx.x;                    // b*836 + i
    const int i = row % NROWS2, bb = row / NROWS2;
    const int src = (i < LT) ? i : (LT + topk_idx[bb * KEEP + (i - LT)]);
    const float* s = x1 + (long long)(bb * NTOK + src) * DIMC;
    float* d = xo + (long long)row * DIMC;
    for (int c = threadIdx.x; c < DIMC; c += 256) d[c] = s[c];
}

__global__ void gelu_bf16(const float* __restrict__ h, bf16_t* __restrict__ o,
                          long long n)
{
    long long t = (long long)blockIdx.x * 256 + threadIdx.x;
    if (t >= n) return;
    float v = h[t];
    o[t] = (bf16_t)(0.5f * v * (1.f + erff(v * 0.70710678118f)));   // exact GELU
}

// ---------------------------------------------------------------------------
extern "C" void kernel_launch(void* const* d_in, const int* in_sizes, int n_in,
                              void* d_out, int out_size, void* d_ws, size_t ws_size,
                              hipStream_t stream)
{
    const float* x        = (const float*)d_in[0];
    const int*   git      = (const int*)  d_in[1];
    const int*   gis      = (const int*)  d_in[2];
    const float* n1w      = (const float*)d_in[3];
    const float* n1b      = (const float*)d_in[4];
    const float* qkv_w    = (const float*)d_in[5];
    const float* proj_w   = (const float*)d_in[6];
    const float* proj_b   = (const float*)d_in[7];
    const float* n2w      = (const float*)d_in[8];
    const float* n2b      = (const float*)d_in[9];
    const float* fc1_w    = (const float*)d_in[10];
    const float* fc1_b    = (const float*)d_in[11];
    const float* fc2_w    = (const float*)d_in[12];
    const float* fc2_b    = (const float*)d_in[13];

    const long long M1 = (long long)BATCH * NTOK;     // 8064
    const long long M2 = (long long)BATCH * NROWS2;   // 6688
    const long long BH = (long long)BATCH * HEADS;    // 96

    // ---- d_out layout (elements, return-tuple order) ----
    float* out_x   = (float*)d_out;                               // 8*836*768
    long long o    = M2 * DIMC;
    int*   out_git = (int*)((float*)d_out + o);  o += BATCH * LT; // 3456
    int*   out_kp  = (int*)((float*)d_out + o);  o += BATCH * KEEP;
    int*   out_rm  = (int*)((float*)d_out + o);  o += BATCH * (LS - KEEP);
    float* out_ce  = (float*)d_out + o;                           // 8*12*576*1008

    // ---- workspace carve (256B aligned) ----
    size_t off = 0;
    auto carve = [&](size_t bytes) {
        size_t p = off; off = (off + bytes + 255) & ~(size_t)255; return p;
    };
    char* ws = (char*)d_ws;
    bf16_t* wq   = (bf16_t*)(ws + carve((size_t)3 * DIMC * DIMC * 2));
    bf16_t* wp   = (bf16_t*)(ws + carve((size_t)DIMC * DIMC * 2));
    bf16_t* w1   = (bf16_t*)(ws + carve((size_t)HIDDEN * DIMC * 2));
    bf16_t* w2   = (bf16_t*)(ws + carve((size_t)DIMC * HIDDEN * 2));
    bf16_t* xn   = (bf16_t*)(ws + carve((size_t)M1 * DIMC * 2));
    float*  qkvf = (float*) (ws + carve((size_t)M1 * 3 * DIMC * 4));
    bf16_t* Qb   = (bf16_t*)(ws + carve((size_t)BH * NTOK * HD * 2));
    bf16_t* Kb   = (bf16_t*)(ws + carve((size_t)BH * NPAD * HD * 2));
    bf16_t* Vt   = (bf16_t*)(ws + carve((size_t)BH * HD * NPAD * 2));
    float*  Sf   = (float*) (ws + carve((size_t)BH * NTOK * NPAD * 4));
    bf16_t* Pb   = (bf16_t*)(ws + carve((size_t)BH * NTOK * NPAD * 2));
    float*  Of   = (float*) (ws + carve((size_t)BH * NTOK * HD * 4));
    bf16_t* ao   = (bf16_t*)(ws + carve((size_t)M1 * DIMC * 2));
    float*  pf   = (float*) (ws + carve((size_t)M1 * DIMC * 4));
    float*  x1   = (float*) (ws + carve((size_t)M1 * DIMC * 4));
    float*  tnb  = (float*) (ws + carve((size_t)BATCH * LS * 4));
    int*    tki  = (int*)   (ws + carve((size_t)BATCH * KEEP * 4));
    bf16_t* xn2  = (bf16_t*)(ws + carve((size_t)M2 * DIMC * 2));
    float*  hf   = (float*) (ws + carve((size_t)M2 * HIDDEN * 4));
    bf16_t* hb   = (bf16_t*)(ws + carve((size_t)M2 * HIDDEN * 2));
    float*  mf   = (float*) (ws + carve((size_t)M2 * DIMC * 4));

    auto blks   = [](long long n) { return (unsigned)((n + 255) / 256); };
    auto tilesN = [](long long n) { return (unsigned)(n / 64); };           // exact
    auto tilesM = [](long long m) { return (unsigned)((m + 127) / 128); };  // tail-masked

    // 1) weights -> bf16
    f32_to_bf16<<<blks(3LL*DIMC*DIMC), 256, 0, stream>>>(qkv_w, wq, 3LL*DIMC*DIMC);
    f32_to_bf16<<<blks((long long)DIMC*DIMC), 256, 0, stream>>>(proj_w, wp, (long long)DIMC*DIMC);
    f32_to_bf16<<<blks((long long)HIDDEN*DIMC), 256, 0, stream>>>(fc1_w, w1, (long long)HIDDEN*DIMC);
    f32_to_bf16<<<blks((long long)DIMC*HIDDEN), 256, 0, stream>>>(fc2_w, w2, (long long)DIMC*HIDDEN);

    // 2) LN1
    layernorm_bf16<<<(unsigned)M1, 256, 0, stream>>>(x, n1w, n1b, xn);

    // 3) QKV GEMM: [8064,768] x [2304,768]^T
    wmma_gemm_bf16<<<dim3(tilesN(3*DIMC), tilesM(M1), 1), 128, 0, stream>>>(
        xn, wq, qkvf, nullptr, (int)M1, DIMC, DIMC, DIMC, 3*DIMC, 0, 0, 0);

    // 4) split heads (K padded to 1024 rows, zero-filled)
    reshape_qk<<<blks(BH*NPAD*HD), 256, 0, stream>>>(qkvf, Qb, Kb, BH*NPAD*HD);
    reshape_vt<<<blks(BH*HD*NPAD), 256, 0, stream>>>(qkvf, Vt, BH*HD*NPAD);

    // 5) scores: per (b,h) [1008,64] x [1024,64]^T -> Sf [1008,1024], batch 96
    wmma_gemm_bf16<<<dim3(tilesN(NPAD), tilesM(NTOK), (unsigned)BH), 128, 0, stream>>>(
        Qb, Kb, Sf, nullptr, NTOK, HD, HD, HD, NPAD,
        (long long)NTOK*HD, (long long)NPAD*HD, (long long)NTOK*NPAD);

    // 6) softmax (+ emit attn_ce slice)
    softmax_attn<<<(unsigned)(BH*NTOK), 256, 0, stream>>>(Sf, Pb, out_ce);

    // 7) attn @ V: per (b,h) [1008,1024] x [64,1024]^T, batch 96
    wmma_gemm_bf16<<<dim3(tilesN(HD), tilesM(NTOK), (unsigned)BH), 128, 0, stream>>>(
        Pb, Vt, Of, nullptr, NTOK, NPAD, NPAD, NPAD, HD,
        (long long)NTOK*NPAD, (long long)HD*NPAD, (long long)NTOK*HD);

    // 8) merge heads -> bf16
    reshape_o<<<blks(M1*DIMC), 256, 0, stream>>>(Of, ao, M1*DIMC);

    // 9) proj GEMM (+bias), then residual x1 = x + proj
    wmma_gemm_bf16<<<dim3(tilesN(DIMC), tilesM(M1), 1), 128, 0, stream>>>(
        ao, wp, pf, proj_b, (int)M1, DIMC, DIMC, DIMC, DIMC, 0, 0, 0);
    add_res<<<blks(M1*DIMC), 256, 0, stream>>>(x, pf, x1, M1*DIMC);

    // 10) token pruning
    attn_norm<<<(unsigned)(BATCH*LS), 256, 0, stream>>>(out_ce, tnb);
    topk_select<<<BATCH, LS, 0, stream>>>(tnb, gis, git, out_git, out_kp, out_rm, tki);
    gather_tokens<<<(unsigned)M2, 256, 0, stream>>>(x1, tki, out_x);

    // 11) LN2 + MLP
    layernorm_bf16<<<(unsigned)M2, 256, 0, stream>>>(out_x, n2w, n2b, xn2);
    wmma_gemm_bf16<<<dim3(tilesN(HIDDEN), tilesM(M2), 1), 128, 0, stream>>>(
        xn2, w1, hf, fc1_b, (int)M2, DIMC, DIMC, DIMC, HIDDEN, 0, 0, 0);
    gelu_bf16<<<blks(M2*HIDDEN), 256, 0, stream>>>(hf, hb, M2*HIDDEN);
    wmma_gemm_bf16<<<dim3(tilesN(DIMC), tilesM(M2), 1), 128, 0, stream>>>(
        hb, w2, mf, fc2_b, (int)M2, HIDDEN, HIDDEN, HIDDEN, DIMC, 0, 0, 0);
    add_inplace<<<blks(M2*DIMC), 256, 0, stream>>>(out_x, mf, M2*DIMC);

    (void)in_sizes; (void)n_in; (void)out_size; (void)ws_size;
}
